// GenesisRNN_5068061409919
// MI455X (gfx1250) — matrix operand order repaired and checked by reading, not compiled, not run on hardware
//
#include <hip/hip_runtime.h>
#include <hip/hip_bf16.h>

typedef __attribute__((ext_vector_type(16))) _Float16 v16h;
typedef __attribute__((ext_vector_type(8)))  _Float16 v8h;
typedef __attribute__((ext_vector_type(8)))  float    v8f;

#define ND      4       // first ND hidden units unclamped
#define GAMMA_C 0.5f
#define WPB     4       // waves per block (wave32)

// Hardware tanh (v_tanh_f32 on gfx1250) with a branch-free fallback.
#if defined(__has_builtin)
#if __has_builtin(__builtin_amdgcn_tanhf)
#define HAVE_HW_TANH 1
#endif
#endif
__device__ __forceinline__ float fast_tanh(float x) {
#ifdef HAVE_HW_TANH
    return __builtin_amdgcn_tanhf(x);
#else
    // branch-free: tanh(x) = (e^{2x}-1)/(e^{2x}+1), v_exp + fast divide
    float e = __expf(2.0f * x);
    return __fdividef(e - 1.0f, e + 1.0f);
#endif
}

// ---------------------------------------------------------------------------
// x [B,S] -> xT [S,B] tiled transpose so the RNN kernel reads x coalesced.
// ---------------------------------------------------------------------------
__global__ void xpose_kernel(const float* __restrict__ x, float* __restrict__ xT,
                             int B, int S) {
    __shared__ float tile[32][33];
    const int s0 = blockIdx.x * 32;
    const int b0 = blockIdx.y * 32;
    const int tx = threadIdx.x, ty = threadIdx.y;
    #pragma unroll
    for (int i = ty; i < 32; i += 8) {
        int bb = b0 + i, ss = s0 + tx;
        if (bb < B && ss < S) tile[i][tx] = x[(size_t)bb * S + ss];
    }
    __syncthreads();
    #pragma unroll
    for (int i = ty; i < 32; i += 8) {
        int ss = s0 + i, bb = b0 + tx;
        if (ss < S && bb < B) xT[(size_t)ss * B + bb] = tile[tx][i];
    }
}

// Padded recurrent weight: rows 0..H-1 = W_hh, row H = W_ih (x slot),
// row H+1 = bias (1.0 slot), rest zero. Columns >= H are zero.
__device__ __forceinline__ float wpad(const float* W_hh, const float* W_ih,
                                      const float* bvec, int H, int k, int j) {
    if (j >= H) return 0.0f;
    if (k < H)  return W_hh[k * H + j];
    if (k == H)     return W_ih[j];
    if (k == H + 1) return bvec[j];
    return 0.0f;
}

// ---------------------------------------------------------------------------
// One wave per 16 batch rows. Hidden state kept in LDS as fp16 [16 x 32];
// each step: store x_t into K-slot H, two ds_load_b128 build the A matrix,
// two v_wmma_f32_16x16x32_f16 produce the pre-activation (affine terms folded
// into K via the x/1.0 slots), hw tanh + clamp (branch-free cndmask),
// unconditional writeback (dead column 31 absorbs out-of-range lanes).
// Single-wave ownership of the tile -> DS in-order, no barriers in the loop.
// ---------------------------------------------------------------------------
__global__ void __launch_bounds__(WPB * 32) rnn_wmma_kernel(
        const float* __restrict__ x, const float* __restrict__ xT, int use_T,
        const float* __restrict__ W_ih, const float* __restrict__ W_hh,
        const float* __restrict__ bvec, const float* __restrict__ fc_w,
        const float* __restrict__ fc_b, float* __restrict__ out,
        int B, int S, int H)
{
    const int lane  = threadIdx.x & 31;
    const int wv    = threadIdx.x >> 5;
    const int mbase = (blockIdx.x * WPB + wv) * 16;
    if (mbase >= B) return;

    __shared__ __align__(16) _Float16 hbuf[WPB][16][32];

    // h0 = 0; pad columns zero; bias column (H+1) = 1.0
    {
        _Float16* hb = &hbuf[wv][0][0];
        for (int i = lane; i < 16 * 32; i += 32) hb[i] = (_Float16)0.0f;
        if (lane < 16) hbuf[wv][lane][H + 1] = (_Float16)1.0f;
    }

    const int hi16 = (lane < 16) ? 0 : 16;   // K-half owned by this lane group
    const int col  = lane & 15;              // N (output column) for B/C/D

    // Recurrent weights in 16-bit B-matrix layout (32 x 16 per tile):
    // element e of lane L holds W[K = (e<8 ? e : e+8) + (L<16 ? 0 : 8)][N].
    v16h Bt0, Bt1;
    #pragma unroll
    for (int e = 0; e < 16; ++e) {
        int k = (e < 8 ? e : e + 8) + (hi16 >> 1);
        Bt0[e] = (_Float16)wpad(W_hh, W_ih, bvec, H, k, col);
        Bt1[e] = (_Float16)wpad(W_hh, W_ih, bvec, H, k, 16 + col);
    }

    v8f czero;
    #pragma unroll
    for (int i = 0; i < 8; ++i) czero[i] = 0.0f;

    const int  row   = col;                       // A-matrix M row for this lane
    const int  loff  = (lane < 16) ? 0 : 8;       // A layout K offsets
    const int  Mrow  = (lane < 16) ? 0 : 8;       // D layout M base
    const bool xlane = (lane < 16) && (mbase + lane < B);
    const bool clamp0 = (col >= ND);              // tile0 clamp predicate (cndmask)
    // out-of-range columns -> dead pad column 31 (zero weight, never read back)
    const int  c0 = (col < H)      ? col      : 31;
    const int  c1 = (16 + col < H) ? 16 + col : 31;

    for (int t = 0; t < S; ++t) {
        // scalar input for this step -> K-slot H
        if (xlane) {
            float xv = use_T ? xT[(size_t)t * B + mbase + lane]
                             : x[(size_t)(mbase + lane) * S + t];
            hbuf[wv][lane][H] = (_Float16)xv;
            if (use_T && t + 8 < S)
                __builtin_prefetch(&xT[(size_t)(t + 8) * B + mbase + lane], 0, 0);
        }

        // A matrix (16x32 fp16): lanes 0-15 K{0-7,16-23}, lanes 16-31 K{8-15,24-31}
        v8h lo = *reinterpret_cast<const v8h*>(&hbuf[wv][row][loff]);
        v8h hi = *reinterpret_cast<const v8h*>(&hbuf[wv][row][16 + loff]);
        v16h A;
        #pragma unroll
        for (int i = 0; i < 8; ++i) { A[i] = lo[i]; A[i + 8] = hi[i]; }

        v8f acc0 = __builtin_amdgcn_wmma_f32_16x16x32_f16(
            false, A, false, Bt0, (short)0, czero, false, false);
        v8f acc1 = __builtin_amdgcn_wmma_f32_16x16x32_f16(
            false, A, false, Bt1, (short)0, czero, false, false);

        // activation + clamp (branch-free), scatter D layout into the LDS state
        #pragma unroll
        for (int r = 0; r < 8; ++r) {
            float h0  = fast_tanh(acc0[r]);
            float h0c = fminf(GAMMA_C, fmaxf(-GAMMA_C, h0));   // v_med3
            h0 = clamp0 ? h0c : h0;                            // v_cndmask
            hbuf[wv][Mrow + r][c0] = (_Float16)h0;

            float h1 = fast_tanh(acc1[r]);                     // N = 16+col >= ND
            h1 = fminf(GAMMA_C, fmaxf(-GAMMA_C, h1));
            hbuf[wv][Mrow + r][c1] = (_Float16)h1;
        }
    }

    // out[b] = h . fc_w + fc_b
    if (xlane) {
        float s = fc_b[0];
        for (int j = 0; j < H; ++j)
            s += (float)hbuf[wv][lane][j] * fc_w[j];
        out[mbase + lane] = s;
    }
}

// ---------------------------------------------------------------------------
extern "C" void kernel_launch(void* const* d_in, const int* in_sizes, int n_in,
                              void* d_out, int out_size, void* d_ws, size_t ws_size,
                              hipStream_t stream) {
    const float* x    = (const float*)d_in[0];
    const float* W_ih = (const float*)d_in[1];
    const float* W_hh = (const float*)d_in[2];
    const float* bvec = (const float*)d_in[3];
    const float* fc_w = (const float*)d_in[4];
    const float* fc_b = (const float*)d_in[5];
    float* out = (float*)d_out;

    const int H = in_sizes[1];           // hidden size (18)
    const int B = out_size;              // batch (4096), output is [B,1]
    const int S = in_sizes[0] / (B > 0 ? B : 1);   // seq len (2048)

    float* xT = (float*)d_ws;
    const int use_T = (ws_size >= (size_t)in_sizes[0] * sizeof(float)) ? 1 : 0;
    if (use_T) {
        dim3 blk(32, 8);
        dim3 grd((S + 31) / 32, (B + 31) / 32);
        hipLaunchKernelGGL(xpose_kernel, grd, blk, 0, stream, x, xT, B, S);
    }

    const int rowsPerBlock = WPB * 16;
    const int nblk = (B + rowsPerBlock - 1) / rowsPerBlock;
    hipLaunchKernelGGL(rnn_wmma_kernel, dim3(nblk), dim3(WPB * 32), 0, stream,
                       x, xT, use_T, W_ih, W_hh, bvec, fc_w, fc_b, out, B, S, H);
}
